// RPN_4964982194866
// MI455X (gfx1250) — matrix operand order repaired and checked by reading, not compiled, not run on hardware
//
#include <hip/hip_runtime.h>
#include <hip/hip_bf16.h>

typedef __attribute__((ext_vector_type(16))) _Float16 v16h;
typedef __attribute__((ext_vector_type(8)))  _Float16 v8h;
typedef __attribute__((ext_vector_type(8)))  float    v8f;

#define H_IMG 64
#define W_IMG 64
#define C_IN  512
#define C_OUT 512
#define M_TOT 4096           // 64*64 spatial
#define K_TOT 4608           // 9*512, K order: k = (kh*3+kw)*512 + c
#define PRE_NMS 12000
#define POST_NMS 1000
#define SORT_N 65536

// chunk permutation (8-half chunks): 0,1,2,3 -> 0,2,1,3  (2-bit swap)
__device__ __forceinline__ int permC(int kc) {
  return ((kc & 1) << 1) | (kc >> 1);
}

// ---------------------------------------------------------------------------
// x (C,H,W) f32 -> xT (H*W, C) f16: spatial-major so an A-tile row is
// contiguous. One-time; writes coalesced.
// ---------------------------------------------------------------------------
__global__ void rpn_cvt_x_T(const float* __restrict__ x,
                            _Float16* __restrict__ xT) {
  int i = blockIdx.x * blockDim.x + threadIdx.x;   // over 512*4096
  if (i >= C_IN * M_TOT) return;
  int s = i >> 9;
  int c = i & 511;
  xT[i] = (_Float16)x[(size_t)c * M_TOT + s];
}

// conv1_w (O,I,3,3) -> wt[n][k'] f16 with k' = p*512 + c  (p = kh*3+kw).
__global__ void rpn_permute_w(const float* __restrict__ w,
                              _Float16* __restrict__ wt) {
  int k = blockIdx.x * blockDim.x + threadIdx.x;   // < 4608
  int n = blockIdx.y;                              // < 512
  int p = k >> 9;
  int c = k & 511;
  wt[(size_t)n * K_TOT + k] = (_Float16)w[(size_t)n * K_TOT + c * 9 + p];
}

// Padded head weights hw[n][k], n in [0,64): 36 locs | 18 scores | pad
__global__ void rpn_build_head_w(const float* __restrict__ locs_w,
                                 const float* __restrict__ scores_w,
                                 _Float16* __restrict__ hw) {
  int i = blockIdx.x * blockDim.x + threadIdx.x;
  if (i >= 64 * C_IN) return;
  int n = i >> 9, k = i & 511;
  float v = 0.0f;
  if (n < 36)      v = locs_w[n * C_IN + k];
  else if (n < 54) v = scores_w[(n - 36) * C_IN + k];
  hw[i] = (_Float16)v;
}

// ---------------------------------------------------------------------------
// Kernel A: 3x3 conv (SAME) + bias + ReLU as implicit-im2col GEMM via WMMA.
// M=4096 spatial, N=512 out-ch, K=4608 (tap-major). yh: (M x 512) f16.
// Block: 256 threads = 8 waves; WG tile 64x64; wave tile 16x32.
// A and B staging are both pure b128 copies (zero-fill at image borders).
// ---------------------------------------------------------------------------
__global__ void rpn_conv3x3_wmma(const _Float16* __restrict__ xT,
                                 const _Float16* __restrict__ wt,
                                 const float* __restrict__ bias,
                                 _Float16* __restrict__ yh) {
  __shared__ _Float16 As[64 * 32];   // M x K tile, chunk-permuted
  __shared__ _Float16 Bs[64 * 32];   // N-major: Bs[n][kk]

  const int tid   = threadIdx.x;
  const int lane  = tid & 31;
  const int wave  = tid >> 5;
  const int lr    = lane & 15;
  const int hi    = lane >> 4;
  const int waveM = wave & 3;
  const int waveN = wave >> 2;
  const int tileM = blockIdx.x * 64;
  const int tileN = blockIdx.y * 64;

  // A-staging: one 8-half chunk per thread per K-step
  const int rA  = tid >> 2;          // 0..63 (tile row)
  const int kcA = tid & 3;           // K chunk within 32
  const int mA  = tileM + rA;
  const int hyB = mA >> 6;
  const int wxB = mA & 63;
  const int dstA = (rA << 2) + permC(kcA);

  v8f c0 = {};
  v8f c1 = {};

  for (int kb = 0; kb < K_TOT; kb += 32) {
    if (kb + 32 < K_TOT)
      __builtin_prefetch(&wt[(size_t)tileN * K_TOT + kb + 32], 0, 1);

    // ---- stage A: tap (kh,kw) constant across this K tile ----
    {
      int p  = kb >> 9;              // 0..8 (scalar)
      int kh = p / 3;
      int kw = p - kh * 3;
      int cb = (kb & 511) >> 3;      // chunk base within channel dim
      int hy = hyB + kh - 1;
      int wx = wxB + kw - 1;
      v8h v = {};
      if (hy >= 0 && hy < H_IMG && wx >= 0 && wx < W_IMG)
        v = ((const v8h*)xT)[(size_t)(hy * W_IMG + wx) * (C_IN / 8) + cb + kcA];
      ((v8h*)As)[dstA] = v;
    }
    // ---- stage B: vector copy, N-major ----
    {
      int n  = tid >> 2;
      int kc = tid & 3;
      ((v8h*)Bs)[(n << 2) + permC(kc)] =
          ((const v8h*)wt)[(size_t)(tileN + n) * (K_TOT / 8) + (kb >> 3) + kc];
    }
    __syncthreads();

    // ---- fragments: one aligned v16h load each (2x ds_load_b128) ----
    v16h a  = *(const v16h*)&As[((waveM * 16 + lr) << 5) + hi * 16];
    v16h b0 = *(const v16h*)&Bs[((waveN * 32 + lr) << 5) + hi * 16];
    v16h b1 = *(const v16h*)&Bs[((waveN * 32 + 16 + lr) << 5) + hi * 16];

    c0 = __builtin_amdgcn_wmma_f32_16x16x32_f16(false, a, false, b0,
                                                (short)0, c0, false, false);
    c1 = __builtin_amdgcn_wmma_f32_16x16x32_f16(false, a, false, b1,
                                                (short)0, c1, false, false);
    __syncthreads();
  }

#pragma unroll
  for (int r = 0; r < 8; ++r) {
    int m  = tileM + waveM * 16 + hi * 8 + r;
    int n0 = tileN + waveN * 32 + lr;
    yh[(size_t)m * C_OUT + n0]      = (_Float16)fmaxf(c0[r] + bias[n0], 0.0f);
    yh[(size_t)m * C_OUT + n0 + 16] = (_Float16)fmaxf(c1[r] + bias[n0 + 16], 0.0f);
  }
}

// ---------------------------------------------------------------------------
// Kernel B: 1x1 heads GEMM: M=4096, K=512, N=64 (36 locs | 18 scores | pad).
// Block: 128 threads = 4 waves; wave -> 16 rows x 64 cols.
// ---------------------------------------------------------------------------
__global__ void rpn_heads_wmma(const _Float16* __restrict__ yh,
                               const _Float16* __restrict__ hw,
                               const float* __restrict__ locs_b,
                               const float* __restrict__ scores_b,
                               float* __restrict__ out_locs,
                               float* __restrict__ out_scores,
                               float* __restrict__ keys) {
  __shared__ _Float16 As[64 * 32];   // chunk-permuted
  __shared__ _Float16 Bs[64 * 32];   // N-major

  const int tid  = threadIdx.x;
  const int lane = tid & 31;
  const int wv   = tid >> 5;
  const int lr   = lane & 15;
  const int hi   = lane >> 4;
  const int tileM = blockIdx.x * 64;

  v8f acc[4];
#pragma unroll
  for (int t = 0; t < 4; ++t) acc[t] = (v8f){};

  for (int kb = 0; kb < C_IN; kb += 32) {
#pragma unroll
    for (int q = 0; q < 2; ++q) {
      int ch = tid + q * 128;
      int r  = ch >> 2;
      int kc = ch & 3;
      ((v8h*)As)[(r << 2) + permC(kc)] =
          ((const v8h*)yh)[(size_t)(tileM + r) * (C_OUT / 8) + (kb >> 3) + kc];
    }
#pragma unroll
    for (int q = 0; q < 2; ++q) {
      int ch = tid + q * 128;
      int n  = ch >> 2;
      int kc = ch & 3;
      ((v8h*)Bs)[(n << 2) + permC(kc)] =
          ((const v8h*)hw)[(size_t)n * (C_IN / 8) + (kb >> 3) + kc];
    }
    __syncthreads();

    v16h a = *(const v16h*)&As[((wv * 16 + lr) << 5) + hi * 16];
#pragma unroll
    for (int t = 0; t < 4; ++t) {
      v16h b = *(const v16h*)&Bs[((t * 16 + lr) << 5) + hi * 16];
      acc[t] = __builtin_amdgcn_wmma_f32_16x16x32_f16(false, a, false, b,
                                                      (short)0, acc[t],
                                                      false, false);
    }
    __syncthreads();
  }

#pragma unroll
  for (int r = 0; r < 8; ++r) {
    int m = tileM + wv * 16 + hi * 8 + r;
#pragma unroll
    for (int t = 0; t < 4; ++t) {
      int n = t * 16 + lr;
      float v = acc[t][r];
      if (n < 36) {
        out_locs[(size_t)m * 36 + n] = v + locs_b[n];
      } else if (n < 54) {
        int j = n - 36;
        float sv = v + scores_b[j];
        out_scores[(size_t)m * 18 + j] = sv;
        if (j & 1) keys[m * 9 + (j >> 1)] = sv;   // fg logit (class 1)
      }
    }
  }
}

// ---------------------------------------------------------------------------
// Anchors + loc2bbox (clip to image).
// ---------------------------------------------------------------------------
__global__ void rpn_anchors_rois(const float* __restrict__ locs,
                                 const int* __restrict__ img_size,
                                 float* __restrict__ anchors,
                                 float* __restrict__ rois) {
  int s = blockIdx.x * blockDim.x + threadIdx.x;
  if (s >= M_TOT * 9) return;
  int a  = s % 9;
  int m  = s / 9;
  int hy = m >> 6;
  int wx = m & 63;
  const float ratios[3] = {0.5f, 1.0f, 2.0f};
  const float scales[3] = {8.0f, 16.0f, 32.0f};
  float r  = ratios[a / 3];
  float sc = scales[a % 3];
  float hA = 16.0f * sc * sqrtf(r);
  float wA = 16.0f * sc * sqrtf(1.0f / r);
  float y1 = hy * 16.0f + 8.0f - 0.5f * hA;
  float x1 = wx * 16.0f + 8.0f - 0.5f * wA;
  float y2 = hy * 16.0f + 8.0f + 0.5f * hA;
  float x2 = wx * 16.0f + 8.0f + 0.5f * wA;
  anchors[s * 4 + 0] = y1;
  anchors[s * 4 + 1] = x1;
  anchors[s * 4 + 2] = y2;
  anchors[s * 4 + 3] = x2;

  float ah = y2 - y1, aw = x2 - x1;
  float acy = y1 + 0.5f * ah, acx = x1 + 0.5f * aw;
  float dy = locs[s * 4 + 0], dx = locs[s * 4 + 1];
  float dh = locs[s * 4 + 2], dw = locs[s * 4 + 3];
  float cy = dy * ah + acy, cx = dx * aw + acx;
  float hh = ah * expf(dh), ww = aw * expf(dw);
  float Hf = (float)img_size[0], Wf = (float)img_size[1];
  rois[s * 4 + 0] = fminf(fmaxf(cy - 0.5f * hh, 0.0f), Hf);
  rois[s * 4 + 1] = fminf(fmaxf(cx - 0.5f * ww, 0.0f), Wf);
  rois[s * 4 + 2] = fminf(fmaxf(cy + 0.5f * hh, 0.0f), Hf);
  rois[s * 4 + 3] = fminf(fmaxf(cx + 0.5f * ww, 0.0f), Wf);
}

// ---------------------------------------------------------------------------
// Exact deterministic top-k: descending bitonic sort of 65536 (key, idx).
// ---------------------------------------------------------------------------
__global__ void rpn_sort_init(float* __restrict__ keys, int* __restrict__ vals) {
  int i = blockIdx.x * blockDim.x + threadIdx.x;
  if (i >= SORT_N) return;
  vals[i] = i;
  if (i >= M_TOT * 9) keys[i] = -3.4e38f;
}

__global__ void rpn_bitonic_pass(float* __restrict__ keys,
                                 int* __restrict__ vals, int j, int k) {
  int i = blockIdx.x * blockDim.x + threadIdx.x;
  int ixj = i ^ j;
  if (ixj <= i) return;
  float ki = keys[i], kj = keys[ixj];
  bool up = ((i & k) == 0);
  bool sw = up ? (ki < kj) : (ki > kj);
  if (sw) {
    keys[i] = kj; keys[ixj] = ki;
    int t = vals[i]; vals[i] = vals[ixj]; vals[ixj] = t;
  }
}

__global__ void rpn_gather_topk(const int* __restrict__ vals,
                                const float* __restrict__ rois,
                                float* __restrict__ roi_tmp) {
  int i = blockIdx.x * blockDim.x + threadIdx.x;
  if (i >= PRE_NMS) return;
  ((float4*)roi_tmp)[i] = ((const float4*)rois)[vals[i]];
}

// ---------------------------------------------------------------------------
// Greedy NMS (sequential over i, block-parallel over j) + ordered compaction.
// ---------------------------------------------------------------------------
__global__ void rpn_nms_compact(const float* __restrict__ b,
                                float* __restrict__ roi_out) {
  __shared__ unsigned char supp[PRE_NMS];
  for (int i = threadIdx.x; i < PRE_NMS; i += blockDim.x) supp[i] = 0;
  __syncthreads();

  for (int i = 0; i < PRE_NMS; ++i) {
    if (!supp[i]) {
      float y1 = b[i * 4 + 0], x1 = b[i * 4 + 1];
      float y2 = b[i * 4 + 2], x2 = b[i * 4 + 3];
      float ai = (y2 - y1) * (x2 - x1);
      for (int j = i + 1 + threadIdx.x; j < PRE_NMS; j += blockDim.x) {
        float by1 = b[j * 4 + 0], bx1 = b[j * 4 + 1];
        float by2 = b[j * 4 + 2], bx2 = b[j * 4 + 3];
        float yy1 = fmaxf(y1, by1), xx1 = fmaxf(x1, bx1);
        float yy2 = fminf(y2, by2), xx2 = fminf(x2, bx2);
        float inter = fmaxf(yy2 - yy1, 0.0f) * fmaxf(xx2 - xx1, 0.0f);
        float aj = (by2 - by1) * (bx2 - bx1);
        float iou = inter / (ai + aj - inter + 1e-9f);
        if (iou > 0.7f) supp[j] = 1;
      }
    }
    __syncthreads();
  }

  if (threadIdx.x == 0) {
    int cnt = 0;
    for (int i = 0; i < PRE_NMS && cnt < POST_NMS; ++i) {
      if (!supp[i]) {
        roi_out[cnt * 4 + 0] = b[i * 4 + 0];
        roi_out[cnt * 4 + 1] = b[i * 4 + 1];
        roi_out[cnt * 4 + 2] = b[i * 4 + 2];
        roi_out[cnt * 4 + 3] = b[i * 4 + 3];
        ++cnt;
      }
    }
    for (; cnt < POST_NMS; ++cnt) {
      roi_out[cnt * 4 + 0] = 0.0f;
      roi_out[cnt * 4 + 1] = 0.0f;
      roi_out[cnt * 4 + 2] = 0.0f;
      roi_out[cnt * 4 + 3] = 0.0f;
    }
  }
}

// ---------------------------------------------------------------------------
extern "C" void kernel_launch(void* const* d_in, const int* in_sizes, int n_in,
                              void* d_out, int out_size, void* d_ws,
                              size_t ws_size, hipStream_t stream) {
  (void)in_sizes; (void)n_in; (void)out_size; (void)ws_size;
  const float* x        = (const float*)d_in[0];
  const float* conv1_w  = (const float*)d_in[1];
  const float* conv1_b  = (const float*)d_in[2];
  const float* locs_w   = (const float*)d_in[3];
  const float* locs_b   = (const float*)d_in[4];
  const float* scores_w = (const float*)d_in[5];
  const float* scores_b = (const float*)d_in[6];
  const int*   img_size = (const int*)d_in[7];

  float* out         = (float*)d_out;
  float* out_locs    = out;                       // 36864*4
  float* out_scores  = out + 147456;              // 36864*2
  float* out_roi     = out + 221184;              // 1000*4
  float* out_anchors = out + 225184;              // 36864*4

  char*      ws      = (char*)d_ws;
  _Float16*  yh      = (_Float16*)(ws);                  // 4096*512 f16 (4 MB)
  _Float16*  xT      = (_Float16*)(ws + 4194304);        // 4096*512 f16 (4 MB)
  _Float16*  wt      = (_Float16*)(ws + 8388608);        // 512*4608 f16 (4.5 MB)
  _Float16*  hw      = (_Float16*)(ws + 13107200);       // 64*512 f16
  float*     keys    = (float*)   (ws + 13172736);       // 65536 f32
  int*       vals    = (int*)     (ws + 13434880);       // 65536 i32
  float*     rois    = (float*)   (ws + 13697024);       // 36864*4 f32
  float*     roi_tmp = (float*)   (ws + 14286848);       // 12000*4 f32

  // 0) one-time f16 pre-conversion / transposition
  rpn_cvt_x_T<<<(C_IN * M_TOT + 255) / 256, 256, 0, stream>>>(x, xT);
  rpn_permute_w<<<dim3(K_TOT / 256, C_OUT), 256, 0, stream>>>(conv1_w, wt);
  rpn_build_head_w<<<(64 * C_IN + 255) / 256, 256, 0, stream>>>(locs_w,
                                                                scores_w, hw);

  // 1) 3x3 conv + ReLU (WMMA)
  rpn_conv3x3_wmma<<<dim3(64, 8), 256, 0, stream>>>(xT, wt, conv1_b, yh);

  // 2) sort buffer init
  rpn_sort_init<<<SORT_N / 256, 256, 0, stream>>>(keys, vals);

  // 3) 1x1 heads (WMMA)
  rpn_heads_wmma<<<64, 128, 0, stream>>>(yh, hw, locs_b, scores_b,
                                         out_locs, out_scores, keys);

  // 4) anchors + loc2bbox
  rpn_anchors_rois<<<(M_TOT * 9 + 255) / 256, 256, 0, stream>>>(
      out_locs, img_size, out_anchors, rois);

  // 5) exact descending bitonic sort
  for (int k = 2; k <= SORT_N; k <<= 1)
    for (int j = k >> 1; j > 0; j >>= 1)
      rpn_bitonic_pass<<<SORT_N / 256, 256, 0, stream>>>(keys, vals, j, k);

  // 6) gather top-12000 boxes in score order
  rpn_gather_topk<<<(PRE_NMS + 255) / 256, 256, 0, stream>>>(vals, rois,
                                                             roi_tmp);

  // 7) greedy NMS + ordered compaction
  rpn_nms_compact<<<1, 512, 0, stream>>>(roi_tmp, out_roi);
}
